// Attend_85555748536825
// MI455X (gfx1250) — matrix operand order repaired and checked by reading, not compiled, not run on hardware
//
#include <hip/hip_runtime.h>
#include <math.h>

// Talking-heads causal attention for MI455X (gfx1250, wave32, WMMA).
// B=4 H=16 N=1024 D=64, fp32 in/out. All four matmuls (QK^T, W_pre mix,
// W_post mix, P@V) run on v_wmma_f32_16x16x32_f16 with fp32 accumulation.

#define BB 4
#define HH 16
#define NN 1024
#define DD 64
#define TQ 16      // query tile (WMMA M)
#define TK 32      // key tile   (WMMA K for f16)
#define STILE (TQ * TK)

typedef __attribute__((ext_vector_type(16))) _Float16 v16h;
typedef __attribute__((ext_vector_type(8)))  float    v8f;

__device__ __forceinline__ float rowmax16(float v) {
  v = fmaxf(v, __shfl_xor(v, 1, 32));
  v = fmaxf(v, __shfl_xor(v, 2, 32));
  v = fmaxf(v, __shfl_xor(v, 4, 32));
  v = fmaxf(v, __shfl_xor(v, 8, 32));
  return v;
}
__device__ __forceinline__ float rowsum16(float v) {
  v += __shfl_xor(v, 1, 32);
  v += __shfl_xor(v, 2, 32);
  v += __shfl_xor(v, 4, 32);
  v += __shfl_xor(v, 8, 32);
  return v;
}

// W (16x16) as an f16 A-fragment, K-dim (input head) zero-padded to 32.
// A layout: lane&15 = output head row, element e -> k = e + (e&8) + 8*half.
__device__ __forceinline__ v16h w_afrag(const float* __restrict__ w,
                                        int l15, int half, float scale) {
  v16h a;
#pragma unroll
  for (int e = 0; e < 16; ++e) {
    const int kk = e + (e & 8) + half * 8;
    a[e] = (kk < HH) ? (_Float16)(w[l15 * HH + kk] * scale) : (_Float16)0.f;
  }
  return a;
}

// One head-mix GEMM over a 32-col slice of one query row:
// C[g, j] = sum_h Wfrag[g,h] * src[h, j].  B layout: lane&15 = column j,
// element e -> h = e + (e&8) + 8*half; padded h (>= 16) supply zeros.
__device__ __forceinline__ void mix_row(const float* __restrict__ s_base,
                                        const v16h& aw, bool bact,
                                        v8f& c0, v8f& c1) {
  v16h b0, b1;
#pragma unroll
  for (int e = 0; e < 16; ++e) {
    const int kk = e + (e & 8) + (bact ? 0 : 8);
    b0[e] = (kk < HH) ? (_Float16)s_base[kk * STILE]      : (_Float16)0.f;
    b1[e] = (kk < HH) ? (_Float16)s_base[kk * STILE + 16] : (_Float16)0.f;
  }
  v8f z = {};
  c0 = __builtin_amdgcn_wmma_f32_16x16x32_f16(false, aw, false, b0, (short)0, z, false, false);
  c1 = __builtin_amdgcn_wmma_f32_16x16x32_f16(false, aw, false, b1, (short)0, z, false, false);
}

// Raw QK^T for one head, one 16x32 score tile -> LDS (fp32 [16][32]).
// K B-fragment gather: lane&15 selects the key row (rows 256B apart across lanes).
__device__ __forceinline__ void qk_tile(const float* __restrict__ kbase, int j0,
                                        const v16h& aq0, const v16h& aq1,
                                        float* __restrict__ sdst, int lane) {
  const int l15 = lane & 15, half = lane >> 4;
#pragma unroll
  for (int jc = 0; jc < 2; ++jc) {
    v16h b0, b1;
#pragma unroll
    for (int e = 0; e < 16; ++e) {
      const float* col = kbase + (size_t)(j0 + jc * 16 + l15) * DD + (e + (e & 8) + half * 8);
      b0[e] = (_Float16)col[0];    // d in [0,32)
      b1[e] = (_Float16)col[32];   // d in [32,64)
    }
    v8f c = {};
    c = __builtin_amdgcn_wmma_f32_16x16x32_f16(false, aq0, false, b0, (short)0, c, false, false);
    c = __builtin_amdgcn_wmma_f32_16x16x32_f16(false, aq1, false, b1, (short)0, c, false, false);
#pragma unroll
    for (int r = 0; r < 8; ++r)
      sdst[(r + 8 * half) * TK + jc * 16 + l15] = c[r];
  }
}

__global__ __launch_bounds__(512, 1)
void talking_heads_attn(const float* __restrict__ q, const float* __restrict__ k,
                        const float* __restrict__ v, const float* __restrict__ wpre,
                        const float* __restrict__ wpost, float* __restrict__ out) {
  __shared__ float s_lds[HH * STILE];  // 32 KB: raw QK^T,   [h][i][j]
  __shared__ float p_lds[HH * STILE];  // 32 KB: norm probs, [h][i][j]
  __shared__ float a_lds[HH * STILE];  // 32 KB: post-mixed, [g][i][j]

  const int i0   = blockIdx.x * TQ;
  const int b    = blockIdx.y;
  const int wv   = threadIdx.x >> 5;   // wave id: owns head wv AND query row wv
  const int lane = threadIdx.x & 31;
  const int l15  = lane & 15, half = lane >> 4;
  const bool bact = (half == 0);       // half 0 supplies h 0..7 in mix B-frags, half 1 h 8..15
  const int gi   = i0 + wv;            // global query row this wave mixes

  const float* qg = q + ((size_t)(b * HH + wv) * NN + i0) * DD;
  const float* kg = k + (size_t)(b * HH + wv) * NN * DD;
  const float* vg = v + (size_t)(b * HH + wv) * NN * DD;

  // Mixing-weight A-fragments (scale folded into W_pre: softmax scale is linear).
  const v16h awpre  = w_afrag(wpre,  l15, half, 0.125f);
  const v16h awpost = w_afrag(wpost, l15, half, 1.0f);

  // Q A-fragments (f16 16x32): lane&15 = row, k = e + (e&8) + 8*half.
  v16h aq0, aq1;
#pragma unroll
  for (int e = 0; e < 16; ++e) {
    const int d = e + (e & 8) + half * 8;
    aq0[e] = (_Float16)qg[l15 * DD + d];
    aq1[e] = (_Float16)qg[l15 * DD + 32 + d];
  }

  const int njt = ((i0 + TQ - 1) >> 5) + 1;   // causal key-tile count
  const float* srow = s_lds + wv * TK + (lane & 15);  // mix B source
  const float* prow = p_lds + wv * TK + (lane & 15);

  // ---------------- pass 1: exact softmax stats (m,l per [g, row wv]) ----
  float m_run[8], l_run[8];
#pragma unroll
  for (int r = 0; r < 8; ++r) { m_run[r] = -3.0e38f; l_run[r] = 0.f; }

  for (int jt = 0; jt < njt; ++jt) {
    const int j0 = jt * TK;
    qk_tile(kg, j0, aq0, aq1, s_lds + wv * STILE, lane);
    __syncthreads();
    v8f c0, c1;                       // c[r] = logits for head g=r+8*half, j=l15
    mix_row(srow, awpre, bact, c0, c1);
#pragma unroll
    for (int r = 0; r < 8; ++r) {
      float s0 = c0[r], s1 = c1[r];
      if (j0 + l15 > gi)      s0 = -3.0e38f;   // causal mask (post-mix)
      if (j0 + 16 + l15 > gi) s1 = -3.0e38f;
      const float rv    = rowmax16(fmaxf(s0, s1));
      const float m_new = fmaxf(m_run[r], rv);
      const float alpha = __expf(m_run[r] - m_new);
      const float p     = __expf(s0 - m_new) + __expf(s1 - m_new);
      l_run[r] = l_run[r] * alpha + rowsum16(p);
      m_run[r] = m_new;
    }
    __syncthreads();
  }

  float rl[8];
#pragma unroll
  for (int r = 0; r < 8; ++r) rl[r] = 1.0f / l_run[r];

  // ---------------- pass 2: probs, post-mix (WMMA), P @ V ----------------
  v8f acc0 = {}, acc1 = {}, acc2 = {}, acc3 = {};   // 16x64 fp32 out tile

  for (int jt = 0; jt < njt; ++jt) {
    const int j0 = jt * TK;
    qk_tile(kg, j0, aq0, aq1, s_lds + wv * STILE, lane);
    __syncthreads();

    // W_pre mix (WMMA) + mask + exact normalization -> p_lds[g][wv][j]
    {
      v8f c0, c1;
      mix_row(srow, awpre, bact, c0, c1);
#pragma unroll
      for (int r = 0; r < 8; ++r) {
        float s0 = c0[r], s1 = c1[r];
        if (j0 + l15 > gi)      s0 = -3.0e38f;
        if (j0 + 16 + l15 > gi) s1 = -3.0e38f;
        float* pd = p_lds + (r + 8 * half) * STILE + wv * TK;
        pd[l15]      = __expf(s0 - m_run[r]) * rl[r];
        pd[16 + l15] = __expf(s1 - m_run[r]) * rl[r];
      }
    }
    __syncthreads();

    // W_post mix (WMMA) -> a_lds[g][wv][j]
    {
      v8f a0, a1;
      mix_row(prow, awpost, bact, a0, a1);
#pragma unroll
      for (int r = 0; r < 8; ++r) {
        float* ad = a_lds + (r + 8 * half) * STILE + wv * TK;
        ad[l15]      = a0[r];
        ad[16 + l15] = a1[r];
      }
    }
    __syncthreads();

    // Head wv: gather its mixed-prob A-fragment and accumulate P @ V.
    v16h pa;
#pragma unroll
    for (int e = 0; e < 16; ++e)
      pa[e] = (_Float16)a_lds[wv * STILE + l15 * TK + (e + (e & 8) + half * 8)];

    // V B-fragments: lane&15 = d column, element e -> j = e + (e&8) + 8*half.
    v16h bv0, bv1, bv2, bv3;
#pragma unroll
    for (int e = 0; e < 16; ++e) {
      const float* vrow = vg + (size_t)(j0 + e + (e & 8) + half * 8) * DD + l15;
      bv0[e] = (_Float16)vrow[0];
      bv1[e] = (_Float16)vrow[16];
      bv2[e] = (_Float16)vrow[32];
      bv3[e] = (_Float16)vrow[48];
    }
    acc0 = __builtin_amdgcn_wmma_f32_16x16x32_f16(false, pa, false, bv0, (short)0, acc0, false, false);
    acc1 = __builtin_amdgcn_wmma_f32_16x16x32_f16(false, pa, false, bv1, (short)0, acc1, false, false);
    acc2 = __builtin_amdgcn_wmma_f32_16x16x32_f16(false, pa, false, bv2, (short)0, acc2, false, false);
    acc3 = __builtin_amdgcn_wmma_f32_16x16x32_f16(false, pa, false, bv3, (short)0, acc3, false, false);
  }

  // Write out[b, wv, i0+row, :] — 64B contiguous per 16-lane chunk.
  float* og = out + ((size_t)(b * HH + wv) * NN + i0) * DD;
#pragma unroll
  for (int r = 0; r < 8; ++r) {
    const int row = r + 8 * half;
    og[row * DD + l15]      = acc0[r];
    og[row * DD + 16 + l15] = acc1[r];
    og[row * DD + 32 + l15] = acc2[r];
    og[row * DD + 48 + l15] = acc3[r];
  }
}

extern "C" void kernel_launch(void* const* d_in, const int* in_sizes, int n_in,
                              void* d_out, int out_size, void* d_ws, size_t ws_size,
                              hipStream_t stream) {
  (void)in_sizes; (void)n_in; (void)out_size; (void)d_ws; (void)ws_size;
  const float* q     = (const float*)d_in[0];
  const float* k     = (const float*)d_in[1];
  const float* v     = (const float*)d_in[2];
  const float* wpre  = (const float*)d_in[3];
  const float* wpost = (const float*)d_in[4];
  float* out = (float*)d_out;
  dim3 grid(NN / TQ, BB);   // 64 query tiles x 4 batches
  dim3 block(HH * 32);      // 16 wave32s: wave w = head w + query-row w
  hipLaunchKernelGGL(talking_heads_attn, grid, block, 0, stream,
                     q, k, v, wpre, wpost, out);
}
